// DGCNNSegmentation_66889820668132
// MI455X (gfx1250) — compile-verified, matrix operand order checked by reading, not run on hardware
//
#include <hip/hip_runtime.h>

typedef __attribute__((ext_vector_type(16))) _Float16 v16h;
typedef __attribute__((ext_vector_type(8)))  _Float16 v8h;
typedef __attribute__((ext_vector_type(8)))  float    v8f;

#define BN_EPS 1e-5f
#define KNN_K  20

// ---------------------------------------------------------------------------
// WMMA helpers (gfx1250, wave32).  16x16x32 f16 -> f32.
// A (16x32): lane holds row M=lane&15; its 16 halves sit at K = hi8+{0..7} and
// hi8+16+{0..7} with hi8 = (lane>>4)*8  ==> two contiguous 16B runs.
// B (32x16) mirrors with N=lane&15.  C/D: lane l -> col l&15, VGPR r -> row
// r + ((l>>4)<<3).
// ---------------------------------------------------------------------------
__device__ __forceinline__ v8f wmma_f16(v16h a, v16h b, v8f c) {
  return __builtin_amdgcn_wmma_f32_16x16x32_f16(
      false, a, false, b, (short)0, c, false, false);
}

// rowbase must point at &tile[row][ks*32 + hi8]; emits two 128-bit loads.
__device__ __forceinline__ v16h ld_tile(const _Float16* rowbase) {
  v8h lo = *(const v8h*)(rowbase);
  v8h hi = *(const v8h*)(rowbase + 16);
  v16h r;
#pragma unroll
  for (int i = 0; i < 8; ++i) { r[i] = lo[i]; r[i + 8] = hi[i]; }
  return r;
}

// ---------------------------------------------------------------------------
// Weight conversion: f32 (Cout,Cin) -> f16 (CoutPad,Kpad), zero padded.
// ---------------------------------------------------------------------------
__global__ __launch_bounds__(256)
void wcvt_kernel(const float* __restrict__ W, int Cout, int Cin,
                 int CoutPad, int Kpad, _Float16* __restrict__ out) {
  int i = blockIdx.x * 256 + threadIdx.x;
  if (i >= CoutPad * Kpad) return;
  int o = i / Kpad, k = i % Kpad;
  float v = (o < Cout && k < Cin) ? W[(long)o * Cin + k] : 0.f;
  out[i] = (_Float16)v;
}

// ---------------------------------------------------------------------------
// Squared norms: xx[b][n] = sum_c x[b][c][n]^2
// ---------------------------------------------------------------------------
__global__ __launch_bounds__(256)
void sqnorm_kernel(const float* __restrict__ x, long stride_b, int C, int N,
                   float* __restrict__ xx) {
  int i = blockIdx.x * 256 + threadIdx.x;
  int b = i / N, n = i % N;
  const float* xb = x + (long)b * stride_b;
  float s = 0.f;
  for (int c = 0; c < C; ++c) {
    float v = xb[(long)c * N + n];
    s += v * v;
  }
  xx[(long)b * N + n] = s;
}

// ---------------------------------------------------------------------------
// KNN: pd = 2*x_i.x_j - |x_i|^2 - |x_j|^2, top-20 per row.
// One wave per 16 query rows.  Query panel + 256-column panels staged in LDS
// (coalesced f32 loads, f16 stores); WMMA sweep with register-resident A
// tiles (KSTEPS templated -> fully unrolled, no movrel).
// ---------------------------------------------------------------------------
template <int KSTEPS>
__global__ __launch_bounds__(32)
void knn_kernel(const float* __restrict__ x, long stride_b, int C, int N,
                const float* __restrict__ xx, int* __restrict__ idx_out) {
  constexpr int KP = KSTEPS * 32;
  __shared__ __align__(16) _Float16 sh_a[16 * KP];
  __shared__ __align__(16) _Float16 sh_b[256 * KP];
  __shared__ float pd[16][256];

  const int lane = threadIdx.x;
  const int nt = N / 16;
  const int b  = blockIdx.x / nt;
  const int q0 = (blockIdx.x % nt) * 16;
  const float* xb = x + (long)b * stride_b;
  const int hi8 = ((lane >> 4) & 1) << 3;

  // stage query panel: sh_a[m][k]
  for (int e = lane; e < 16 * KP; e += 32) {
    int m = e & 15, k = e >> 4;
    float v = (k < C) ? xb[(long)k * N + q0 + m] : 0.f;
    sh_a[m * KP + k] = (_Float16)v;
  }
  __syncthreads();

  v16h a_t[KSTEPS];
#pragma unroll
  for (int ks = 0; ks < KSTEPS; ++ks)
    a_t[ks] = ld_tile(sh_a + (lane & 15) * KP + ks * 32 + hi8);

  float best_d[KNN_K];
  int   best_i[KNN_K];
#pragma unroll
  for (int i = 0; i < KNN_K; ++i) { best_d[i] = -3.4e38f; best_i[i] = 0; }

  for (int j0 = 0; j0 < N; j0 += 256) {
    // stage column panel: sh_b[col][k], coalesced over col
    for (int e = lane; e < 256 * KP; e += 32) {
      int col = e & 255, k = e >> 8;
      float v = (k < C) ? xb[(long)k * N + j0 + col] : 0.f;
      sh_b[col * KP + k] = (_Float16)v;
    }
    __syncthreads();

    for (int jt = 0; jt < 256; jt += 16) {
      v8f acc = {};
#pragma unroll
      for (int ks = 0; ks < KSTEPS; ++ks) {
        v16h bt = ld_tile(sh_b + (jt + (lane & 15)) * KP + ks * 32 + hi8);
        acc = wmma_f16(a_t[ks], bt, acc);
      }
      const int bn = j0 + jt + (lane & 15);
      const float xxc = xx[(long)b * N + bn];
#pragma unroll
      for (int r = 0; r < 8; ++r) {
        int m = r + hi8;
        pd[m][jt + (lane & 15)] = 2.f * acc[r] - xx[(long)b * N + q0 + m] - xxc;
      }
    }
    __syncthreads();
    if (lane < 16) {
      for (int cc = 0; cc < 256; ++cc) {
        float d = pd[lane][cc];
        if (d > best_d[KNN_K - 1]) {
          int col = j0 + cc;
          int p = KNN_K - 1;
          while (p > 0 && best_d[p - 1] < d) {
            best_d[p] = best_d[p - 1];
            best_i[p] = best_i[p - 1];
            --p;
          }
          best_d[p] = d;
          best_i[p] = col;
        }
      }
    }
    __syncthreads();
  }
  if (lane < 16) {
    int* o = idx_out + ((long)b * N + q0 + lane) * KNN_K;
#pragma unroll
    for (int i = 0; i < KNN_K; ++i) o[i] = best_i[i];
  }
}

// ---------------------------------------------------------------------------
// EdgeConv: one wave per (b, n).  Gathered [nbr-ctr, ctr] features staged in
// LDS f16 (rows padded to 32), weights pre-converted f16 read as b128 pairs,
// BN + LeakyReLU, masked max over k with cross-half shuffle.
// ---------------------------------------------------------------------------
template <int KSTEPS>   // (2C+31)/32
__global__ __launch_bounds__(32)
void edgeconv_kernel(const float* __restrict__ xin, long in_stride_b, int C, int N,
                     const int* __restrict__ knn_idx,
                     const _Float16* __restrict__ Wh,        // (Cout, KP) f16
                     const float* __restrict__ bn_g, const float* __restrict__ bn_b,
                     const float* __restrict__ bn_m, const float* __restrict__ bn_v,
                     int Cout,
                     float* __restrict__ out, long out_stride_b, int out_choff) {
  constexpr int KP = KSTEPS * 32;
  __shared__ __align__(16) _Float16 feat[32 * KP];
  const int lane = threadIdx.x;
  const int b = blockIdx.x / N;
  const int n = blockIdx.x % N;
  const float* xb = xin + (long)b * in_stride_b;
  const int C2 = 2 * C;
  const int hi8 = ((lane >> 4) & 1) << 3;

  {  // zero-fill (pad rows 20..31 and pad channels must be 0)
    int* fz = (int*)feat;
    for (int e = lane; e < 32 * KP / 2; e += 32) fz[e] = 0;
  }
  __syncthreads();

  const int* idxp = knn_idx + ((long)b * N + n) * KNN_K;
  for (int kk = 0; kk < KNN_K; ++kk) {
    int nbr = idxp[kk];
    for (int c = lane; c < C2; c += 32) {
      float val = (c < C) ? (xb[(long)c * N + nbr] - xb[(long)c * N + n])
                          : xb[(long)(c - C) * N + n];
      feat[kk * KP + c] = (_Float16)val;
    }
  }
  __syncthreads();

  for (int o0 = 0; o0 < Cout; o0 += 16) {
    v8f acc0 = {}, acc1 = {};
    const int o = o0 + (lane & 15);
#pragma unroll
    for (int ks = 0; ks < KSTEPS; ++ks) {
      v16h bt = ld_tile(Wh + (long)o * KP + ks * 32 + hi8);
      v16h a0 = ld_tile(feat + (lane & 15) * KP + ks * 32 + hi8);
      v16h a1 = ld_tile(feat + (16 + (lane & 15)) * KP + ks * 32 + hi8);
      acc0 = wmma_f16(a0, bt, acc0);
      acc1 = wmma_f16(a1, bt, acc1);
    }
    const float sc = bn_g[o] * rsqrtf(bn_v[o] + BN_EPS);
    const float sh = bn_b[o] - sc * bn_m[o];
    float mx = -3.4e38f;
#pragma unroll
    for (int r = 0; r < 8; ++r) {
      float y0 = sc * acc0[r] + sh;
      y0 = (y0 > 0.f) ? y0 : 0.2f * y0;
      mx = fmaxf(mx, y0);
      int kk1 = 16 + r + hi8;
      if (kk1 < KNN_K) {
        float y1 = sc * acc1[r] + sh;
        y1 = (y1 > 0.f) ? y1 : 0.2f * y1;
        mx = fmaxf(mx, y1);
      }
    }
    mx = fmaxf(mx, __shfl_xor(mx, 16, 32));
    if (lane < 16)
      out[(long)b * out_stride_b + (long)(out_choff + o) * N + n] = mx;
  }
}

// ---------------------------------------------------------------------------
// 1x1 MLP: 16 points per wave; input row-block staged in LDS f16 (broadcast
// global-max channels spliced in for c7); weights f16 (CoutPad,Kpad) as b128
// pairs; BN+LReLU or bias epilogue.
// ---------------------------------------------------------------------------
__global__ __launch_bounds__(32)
void mlp_kernel(const float* __restrict__ inA, int C1,   // (B, C1, N)
                const float* __restrict__ inG, int C2,   // (B, C2) broadcast
                const _Float16* __restrict__ Wh,         // (CoutPad, Kpad) f16
                const float* __restrict__ bn_g, const float* __restrict__ bn_b,
                const float* __restrict__ bn_m, const float* __restrict__ bn_v,
                const float* __restrict__ bias,          // when bn_g == null
                int Cout, int N,
                float* __restrict__ outp) {              // (B, Cout, N)
  __shared__ __align__(16) _Float16 arow[16 * 1536];     // 48 KB, Kpad <= 1536
  const int lane = threadIdx.x;
  const int nt = N / 16;
  const int b  = blockIdx.x / nt;
  const int n0 = (blockIdx.x % nt) * 16;
  const int Cin  = C1 + C2;
  const int Kpad = (Cin + 31) & ~31;
  const int hi8 = ((lane >> 4) & 1) << 3;

  for (int e = lane; e < 16 * Kpad; e += 32) {
    int row = e & 15, c = e >> 4;       // coalesced over row
    float val = 0.f;
    if (c < C1)       val = inA[((long)b * C1 + c) * N + n0 + row];
    else if (c < Cin) val = inG[(long)b * C2 + (c - C1)];
    arow[row * Kpad + c] = (_Float16)val;
  }
  __syncthreads();

  const int ksteps = Kpad / 32;
  const int CoutT = (Cout + 15) & ~15;

  for (int o0 = 0; o0 < CoutT; o0 += 16) {
    v8f acc = {};
    const int o = o0 + (lane & 15);     // valid row in padded Wh
    for (int ks = 0; ks < ksteps; ++ks) {
      v16h a  = ld_tile(arow + (lane & 15) * Kpad + ks * 32 + hi8);
      v16h bt = ld_tile(Wh + (long)o * Kpad + ks * 32 + hi8);
      acc = wmma_f16(a, bt, acc);
    }
    if (o < Cout) {
      float sc, sh;
      if (bn_g) {
        sc = bn_g[o] * rsqrtf(bn_v[o] + BN_EPS);
        sh = bn_b[o] - sc * bn_m[o];
      } else {
        sc = 1.f;
        sh = bias[o];
      }
#pragma unroll
      for (int r = 0; r < 8; ++r) {
        int m = r + hi8;
        float y = sc * acc[r] + sh;
        if (bn_g) y = (y > 0.f) ? y : 0.2f * y;
        outp[((long)b * Cout + o) * N + n0 + m] = y;
      }
    }
  }
}

// ---------------------------------------------------------------------------
// Global max over N: one block per (b, c) row.
// ---------------------------------------------------------------------------
__global__ __launch_bounds__(256)
void maxpool_kernel(const float* __restrict__ in, int N, float* __restrict__ outg) {
  __shared__ float red[256];
  const float* p = in + (long)blockIdx.x * N;
  float m = -3.4e38f;
  for (int i = threadIdx.x; i < N; i += 256) m = fmaxf(m, p[i]);
  red[threadIdx.x] = m;
  __syncthreads();
  for (int s = 128; s > 0; s >>= 1) {
    if ((int)threadIdx.x < s)
      red[threadIdx.x] = fmaxf(red[threadIdx.x], red[threadIdx.x + s]);
    __syncthreads();
  }
  if (threadIdx.x == 0) outg[blockIdx.x] = red[0];
}

// ---------------------------------------------------------------------------
// Host driver
// ---------------------------------------------------------------------------
extern "C" void kernel_launch(void* const* d_in, const int* in_sizes, int n_in,
                              void* d_out, int out_size, void* d_ws, size_t ws_size,
                              hipStream_t stream) {
  (void)in_sizes; (void)n_in; (void)out_size; (void)ws_size;
  const int B = 4, N = 4096;

  struct Blk { const float *W, *g, *b, *m, *v; };
  int pi = 1;
  auto nextf = [&]() { return (const float*)d_in[pi++]; };
  auto getBlk = [&]() {
    Blk bl;
    bl.W = nextf(); bl.g = nextf(); bl.b = nextf(); bl.m = nextf(); bl.v = nextf();
    return bl;
  };
  const float* x = (const float*)d_in[0];
  Blk ec1 = getBlk(), ec2 = getBlk(), ec3 = getBlk(), ec4 = getBlk(), ec5 = getBlk();
  Blk c6 = getBlk(), c7 = getBlk(), c8 = getBlk();
  const float* c9W = nextf();
  const float* c9b = nextf();

  // workspace carve
  char* wsp = (char*)d_ws;
  auto carve = [&](size_t bytes) {
    void* p = (void*)wsp;
    wsp += (bytes + 255) & ~(size_t)255;
    return p;
  };
  float* cat   = (float*)carve((size_t)B * 496 * N * 4);
  float* c6o   = (float*)carve((size_t)B * 1024 * N * 4);
  float* gmax  = (float*)carve((size_t)B * 1024 * 4);
  float* c7o   = (float*)carve((size_t)B * 384 * N * 4);
  float* c8o   = (float*)carve((size_t)B * 192 * N * 4);
  float* xx    = (float*)carve((size_t)B * N * 4);
  int*   knn_i = (int*)  carve((size_t)B * N * KNN_K * 4);

  auto carveW = [&](const float* W, int Cout, int Cin, int CoutPad, int Kpad) {
    _Float16* p = (_Float16*)carve((size_t)CoutPad * Kpad * 2);
    int tot = CoutPad * Kpad;
    wcvt_kernel<<<(tot + 255) / 256, 256, 0, stream>>>(W, Cout, Cin, CoutPad, Kpad, p);
    return p;
  };
  _Float16* ec1h = carveW(ec1.W, 64, 6, 64, 32);
  _Float16* ec2h = carveW(ec2.W, 64, 128, 64, 128);
  _Float16* ec3h = carveW(ec3.W, 48, 128, 48, 128);
  _Float16* ec4h = carveW(ec4.W, 128, 96, 128, 96);
  _Float16* ec5h = carveW(ec5.W, 192, 256, 192, 256);
  _Float16* c6h  = carveW(c6.W, 1024, 496, 1024, 512);
  _Float16* c7h  = carveW(c7.W, 384, 1520, 384, 1536);
  _Float16* c8h  = carveW(c8.W, 192, 384, 192, 384);
  _Float16* c9h  = carveW(c9W, 5, 192, 16, 192);

  const long cat_sb = (long)496 * N;

  auto edge = [&](const float* in, long in_sb, int C, Blk& p, _Float16* Wh,
                  int Cout, int choff) {
    sqnorm_kernel<<<(B * N) / 256, 256, 0, stream>>>(in, in_sb, C, N, xx);
    const int gk = B * (N / 16);
    switch ((C + 31) / 32) {
      case 1: knn_kernel<1><<<gk, 32, 0, stream>>>(in, in_sb, C, N, xx, knn_i); break;
      case 2: knn_kernel<2><<<gk, 32, 0, stream>>>(in, in_sb, C, N, xx, knn_i); break;
      default: knn_kernel<4><<<gk, 32, 0, stream>>>(in, in_sb, C, N, xx, knn_i); break;
    }
    const int ge = B * N;
    switch ((2 * C + 31) / 32) {
      case 1: edgeconv_kernel<1><<<ge, 32, 0, stream>>>(in, in_sb, C, N, knn_i,
                  Wh, p.g, p.b, p.m, p.v, Cout, cat, cat_sb, choff); break;
      case 3: edgeconv_kernel<3><<<ge, 32, 0, stream>>>(in, in_sb, C, N, knn_i,
                  Wh, p.g, p.b, p.m, p.v, Cout, cat, cat_sb, choff); break;
      case 4: edgeconv_kernel<4><<<ge, 32, 0, stream>>>(in, in_sb, C, N, knn_i,
                  Wh, p.g, p.b, p.m, p.v, Cout, cat, cat_sb, choff); break;
      default: edgeconv_kernel<8><<<ge, 32, 0, stream>>>(in, in_sb, C, N, knn_i,
                  Wh, p.g, p.b, p.m, p.v, Cout, cat, cat_sb, choff); break;
    }
  };

  // 5 EdgeConv layers, outputs written into concat buffer at channel offsets
  edge(x,                   (long)3 * N,   3, ec1, ec1h,  64,   0);
  edge(cat + (long)0   * N, cat_sb,       64, ec2, ec2h,  64,  64);
  edge(cat + (long)64  * N, cat_sb,       64, ec3, ec3h,  48, 128);
  edge(cat + (long)128 * N, cat_sb,       48, ec4, ec4h, 128, 176);
  edge(cat + (long)176 * N, cat_sb,      128, ec5, ec5h, 192, 304);

  const int grid16 = B * (N / 16);
  // c6: 496 -> 1024, then global max pool
  mlp_kernel<<<grid16, 32, 0, stream>>>(cat, 496, nullptr, 0, c6h,
                                        c6.g, c6.b, c6.m, c6.v, nullptr,
                                        1024, N, c6o);
  maxpool_kernel<<<B * 1024, 256, 0, stream>>>(c6o, N, gmax);
  // c7: [cat(496) | gmax(1024 broadcast)] -> 384
  mlp_kernel<<<grid16, 32, 0, stream>>>(cat, 496, gmax, 1024, c7h,
                                        c7.g, c7.b, c7.m, c7.v, nullptr,
                                        384, N, c7o);
  // c8: 384 -> 192
  mlp_kernel<<<grid16, 32, 0, stream>>>(c7o, 384, nullptr, 0, c8h,
                                        c8.g, c8.b, c8.m, c8.v, nullptr,
                                        192, N, c8o);
  // c9: 192 -> 5 (+bias), straight to d_out (B,5,N)
  mlp_kernel<<<grid16, 32, 0, stream>>>(c8o, 192, nullptr, 0, c9h,
                                        nullptr, nullptr, nullptr, nullptr, c9b,
                                        5, N, (float*)d_out);
}
